// GCN_1623497638183
// MI455X (gfx1250) — compile-verified, matrix operand order checked by reading, not run on hardware
//
#include <hip/hip_runtime.h>
#include <hip/hip_bf16.h>

// ---------------------------------------------------------------------------
// GCN 3-layer forward for MI455X (gfx1250, wave32).
//   h1 = relu( scatter_add((x @W1)[src] -> dst) + b1 )
//   h2 = relu( scatter_add((h1@W2)[src] -> dst) + b2 )
//   out =      scatter_add((h2@W3)[src] -> dst) + b3
// GEMMs use v_wmma_f32_16x16x32_bf16 (bf16 in, f32 accumulate).
// Scatter-add uses global_atomic_add_f32; agg + T buffers (25.6MB each) are
// L2-resident (192MB L2), so the random edge traffic never hits HBM.
// ---------------------------------------------------------------------------

#define N_NODES 50000
#define N_EDGES 800000
#define IN_CH   128
#define HID_CH  128
#define OUT_CH  64

typedef __attribute__((ext_vector_type(16))) __bf16 v16bf;
typedef __attribute__((ext_vector_type(8)))  float  v8f;

// --- transpose + convert f32 weights [K][N] -> bf16 [N][K] ------------------
__global__ void wt_convert_kernel(const float* __restrict__ W,
                                  __bf16* __restrict__ Wt, int K, int N) {
    int i = blockIdx.x * blockDim.x + threadIdx.x;
    if (i >= K * N) return;
    int k = i / N, n = i % N;
    Wt[(size_t)n * K + k] = (__bf16)W[(size_t)k * N + n];
}

// --- init aggregation buffer to broadcast bias ------------------------------
template<int C>
__global__ void init_bias_kernel(float* __restrict__ agg,
                                 const float* __restrict__ b, int total) {
    int i = blockIdx.x * blockDim.x + threadIdx.x;
    if (i < total) agg[i] = b[i & (C - 1)];
}

// --- WMMA GEMM: T[M,N] = (relu?)(H[M,K]) @ Wt^T  (Wt is bf16 [N][K]) --------
// One wave per 16x16 output tile; K consumed 32/step via wmma_f32_16x16x32_bf16.
// A fragment layout (16-bit 16x32, ISA 7.12.2): lane L holds row M=L%16,
//   kbase=(L>=16)*8; vector elems 0..7 -> K=kbase+0..7, 8..15 -> K=kbase+16..23.
// B fragment mirrors A with the column index in place of the row (Wt is
// stored column-major-of-W so loads are contiguous like A).
template<int KDIM, int NDIM, bool RELU>
__global__ void gemm_wmma_kernel(const float* __restrict__ H,
                                 const __bf16* __restrict__ Wt,
                                 float* __restrict__ T, int M) {
    const int lane = threadIdx.x & 31;
    const int wave = (blockIdx.x * blockDim.x + threadIdx.x) >> 5;
    constexpr int NT = NDIM / 16;
    const int mt = wave / NT;
    const int nt = wave % NT;
    if (mt * 16 >= M) return;               // wave-uniform: EXEC stays all-1s

    const int arow  = mt * 16 + (lane & 15);
    const int bcol  = nt * 16 + (lane & 15);
    const int kbase = (lane >> 4) * 8;

    v8f acc = {};
    #pragma unroll
    for (int k0 = 0; k0 < KDIM; k0 += 32) {
        const float*  ap = H  + (size_t)arow * KDIM + k0 + kbase;
        const __bf16* bp = Wt + (size_t)bcol * KDIM + k0 + kbase;
        v16bf a, b;
        #pragma unroll
        for (int j = 0; j < 8; ++j) {
            float f0 = ap[j];
            float f1 = ap[16 + j];
            if (RELU) { f0 = fmaxf(f0, 0.0f); f1 = fmaxf(f1, 0.0f); }
            a[j]     = (__bf16)f0;
            a[8 + j] = (__bf16)f1;
            b[j]     = bp[j];
            b[8 + j] = bp[16 + j];
        }
        acc = __builtin_amdgcn_wmma_f32_16x16x32_bf16(
                  /*neg_a=*/false, a, /*neg_b=*/false, b,
                  /*c_mod=*/(short)0, acc, /*reuse_a=*/false, /*reuse_b=*/false);
    }

    // C/D layout: lanes 0-15 -> rows 0..7, lanes 16-31 -> rows 8..15
    const int crow0 = mt * 16 + (lane >> 4) * 8;
    const int ccol  = nt * 16 + (lane & 15);
    #pragma unroll
    for (int r = 0; r < 8; ++r)
        T[(size_t)(crow0 + r) * NDIM + ccol] = acc[r];
}

// --- edge scatter-add: agg[dst] += T[src], one wave per edge ----------------
// Lane l handles VPL consecutive channels; loads vectorize (b128/b64) and the
// VPL atomic f32 adds stay inside L2.
template<int C, int VPL>
__global__ void scatter_add_kernel(const long long* __restrict__ ei,
                                   const float* __restrict__ T,
                                   float* __restrict__ agg, int nE) {
    constexpr int LPE = C / VPL;            // lanes per edge (=32)
    int gid = blockIdx.x * blockDim.x + threadIdx.x;
    int e = gid / LPE;
    if (e >= nE) return;
    int l = gid % LPE;
    int s = (int)ei[e];
    int d = (int)ei[(size_t)nE + e];
    const float* sp = T   + (size_t)s * C + l * VPL;
    float*       dp = agg + (size_t)d * C + l * VPL;
    float v[VPL];
    #pragma unroll
    for (int i = 0; i < VPL; ++i) v[i] = sp[i];
    #pragma unroll
    for (int i = 0; i < VPL; ++i) atomicAdd(dp + i, v[i]);
}

extern "C" void kernel_launch(void* const* d_in, const int* in_sizes, int n_in,
                              void* d_out, int out_size, void* d_ws, size_t ws_size,
                              hipStream_t stream) {
    const float*     x  = (const float*)d_in[0];
    const long long* ei = (const long long*)d_in[1];   // int64 edge_index [2,E]
    const float*     W1 = (const float*)d_in[2];
    const float*     b1 = (const float*)d_in[3];
    const float*     W2 = (const float*)d_in[4];
    const float*     b2 = (const float*)d_in[5];
    const float*     W3 = (const float*)d_in[6];
    const float*     b3 = (const float*)d_in[7];
    float* out = (float*)d_out;

    // workspace carve-up (~51.3 MB)
    char* ws = (char*)d_ws;
    __bf16* Wt1 = (__bf16*)(ws);                        // 128*128*2 = 32768 B
    __bf16* Wt2 = (__bf16*)(ws + 32768);                // 32768 B
    __bf16* Wt3 = (__bf16*)(ws + 65536);                // 128*64*2 = 16384 B
    float*  bufA = (float*)(ws + 81920);                // 50000*128*4 = 25.6 MB
    float*  bufB = bufA + (size_t)N_NODES * HID_CH;     // 25.6 MB

    const int blk = 256;

    // weights -> bf16 [N][K]
    wt_convert_kernel<<<(IN_CH * HID_CH + blk - 1) / blk, blk, 0, stream>>>(W1, Wt1, IN_CH, HID_CH);
    wt_convert_kernel<<<(HID_CH * HID_CH + blk - 1) / blk, blk, 0, stream>>>(W2, Wt2, HID_CH, HID_CH);
    wt_convert_kernel<<<(HID_CH * OUT_CH + blk - 1) / blk, blk, 0, stream>>>(W3, Wt3, HID_CH, OUT_CH);

    const int mtiles = N_NODES / 16;                    // 3125 (exact)
    const int waves128 = mtiles * (HID_CH / 16);        // 25000
    const int waves64  = mtiles * (OUT_CH / 16);        // 12500
    const int gblk128 = (waves128 * 32 + blk - 1) / blk;  // 3125 blocks
    const int gblk64  = (waves64  * 32 + blk - 1) / blk;  // 1563 blocks
    const int sblk128 = (N_EDGES * (HID_CH / 4) + blk - 1) / blk;  // 100000
    const int sblk64  = (N_EDGES * (OUT_CH / 2) + blk - 1) / blk;  // 100000

    // ---- layer 1: T = x@W1 ; agg = b1 ; agg += T[src]->dst ----
    gemm_wmma_kernel<IN_CH, HID_CH, false><<<gblk128, blk, 0, stream>>>(x, Wt1, bufA, N_NODES);
    init_bias_kernel<HID_CH><<<(N_NODES * HID_CH + blk - 1) / blk, blk, 0, stream>>>(bufB, b1, N_NODES * HID_CH);
    scatter_add_kernel<HID_CH, 4><<<sblk128, blk, 0, stream>>>(ei, bufA, bufB, N_EDGES);

    // ---- layer 2: T = relu(agg1)@W2 ; agg2 = b2 ; agg2 += T[src]->dst ----
    gemm_wmma_kernel<HID_CH, HID_CH, true><<<gblk128, blk, 0, stream>>>(bufB, Wt2, bufA, N_NODES);
    init_bias_kernel<HID_CH><<<(N_NODES * HID_CH + blk - 1) / blk, blk, 0, stream>>>(bufB, b2, N_NODES * HID_CH);
    scatter_add_kernel<HID_CH, 4><<<sblk128, blk, 0, stream>>>(ei, bufA, bufB, N_EDGES);

    // ---- layer 3: T = relu(agg2)@W3 ; out = b3 ; out += T[src]->dst ----
    gemm_wmma_kernel<HID_CH, OUT_CH, true><<<gblk64, blk, 0, stream>>>(bufB, Wt3, bufA, N_NODES);
    init_bias_kernel<OUT_CH><<<(N_NODES * OUT_CH + blk - 1) / blk, blk, 0, stream>>>(out, b3, N_NODES * OUT_CH);
    scatter_add_kernel<OUT_CH, 2><<<sblk64, blk, 0, stream>>>(ei, bufA, out, N_EDGES);
}
